// AttentionMPNNWithEdgeFeatures_65352222376827
// MI455X (gfx1250) — matrix operand order repaired and checked by reading, not compiled, not run on hardware
//
#include <hip/hip_runtime.h>
#include <math.h>

typedef __attribute__((ext_vector_type(16))) _Float16 v16h;
typedef __attribute__((ext_vector_type(8)))  float    v8f;

#define ND  256
#define ED  128
#define CATD 640
#define TILE_E 32

// ---------------------------------------------------------------------------
// Fragment loaders for WMMA 16x16x32 f16: A and B both use contiguous k = 16*(L/16) + s
// ---------------------------------------------------------------------------
union FragH { v16h v; float4 f4[2]; };

// A (16x32, MxK) from LDS row-major [M][ldk]:
//   lane L: M = L&15 ; halves 0..15 -> K = off+0..15
//   where off = kbase + ((L>=16)?16:0)
__device__ __forceinline__ v16h load_a_frag(const _Float16* __restrict__ lds,
                                            int ldk, int mtile, int kbase, int lane) {
  const int m = mtile * 16 + (lane & 15);
  const _Float16* p = lds + (size_t)m * ldk + kbase + ((lane >> 4) << 4);
  FragH u;
  u.f4[0] = *(const float4*)(p);
  u.f4[1] = *(const float4*)(p + 8);
  return u.v;
}

// B (32x16, KxN) from global transposed weights wT[N][Kin]:
//   lane L: N = L&15 ; K = kbase + ((L>=16)?16:0) + h  (16 contiguous halfs)
__device__ __forceinline__ v16h load_b_frag(const _Float16* __restrict__ wT,
                                            int Kin, int ntile, int kbase, int lane) {
  const int n = ntile * 16 + (lane & 15);
  const _Float16* p = wT + (size_t)n * Kin + kbase + ((lane >> 4) << 4);
  FragH u;
  u.f4[0] = *(const float4*)(p);
  u.f4[1] = *(const float4*)(p + 8);
  return u.v;
}

// ---------------------------------------------------------------------------
// One MLP layer over a 32-edge tile, cooperatively by 8 waves.
// MODE 0: relu(out+bias) -> LDS f16   MODE 1: alpha-scatter atomicAdd x_out
// MODE 2: +bias -> e_out global
// ---------------------------------------------------------------------------
template<int MODE>
__device__ __forceinline__ void wave_mlp_layer(
    const _Float16* __restrict__ inLds, int Kin,
    const _Float16* __restrict__ wT, const float* __restrict__ bias, int Nout,
    _Float16* __restrict__ outLds,
    float* __restrict__ x_out, const int* __restrict__ src_l,
    const float* __restrict__ alpha_l,
    float* __restrict__ e_out_tile, int nValid)
{
  const int tid  = threadIdx.x;
  const int lane = tid & 31;
  const int wave = tid >> 5;
  const int nT   = 2 * (Nout >> 4);     // 2 M-tiles x (Nout/16) N-tiles

  for (int t = wave; t < nT; t += 8) {
    const int mtile = t & 1;
    const int ntile = t >> 1;
    v8f acc = {};
    for (int kb = 0; kb < Kin; kb += 32) {
      v16h a = load_a_frag(inLds, Kin, mtile, kb, lane);
      v16h b = load_b_frag(wT, Kin, ntile, kb, lane);
      acc = __builtin_amdgcn_wmma_f32_16x16x32_f16(
          false, a, false, b, (short)0, acc, false, false);
    }
    // C/D layout: lane L -> N = ntile*16 + (L&15); VGPR r -> M = r + ((L>=16)?8:0)
    const int ncol  = ntile * 16 + (lane & 15);
    const int mrow0 = mtile * 16 + ((lane >> 4) << 3);
    const float bv  = bias[ncol];
#pragma unroll
    for (int r = 0; r < 8; ++r) {
      const int m = mrow0 + r;
      float v = acc[r] + bv;
      if (MODE == 0) {
        outLds[(size_t)m * Nout + ncol] = (_Float16)fmaxf(v, 0.0f);
      } else if (MODE == 1) {
        if (m < nValid)
          atomicAdd(&x_out[(size_t)src_l[m] * ND + ncol], alpha_l[m] * v);
      } else {
        if (m < nValid)
          e_out_tile[(size_t)m * ED + ncol] = v;
      }
    }
  }
}

// ---------------------------------------------------------------------------
// Weight convert + transpose: w[K][N] f32 -> wT[N][K] f16
// ---------------------------------------------------------------------------
__global__ void k_wt(const float* __restrict__ w, _Float16* __restrict__ wT,
                     int K, int N) {
  int i = blockIdx.x * blockDim.x + threadIdx.x;
  if (i < K * N) {
    int k = i / N, n = i - k * N;
    wT[(size_t)n * K + k] = (_Float16)w[i];
  }
}

// zero x_out, zero segsum, segmax = -inf
__global__ void k_init(float* __restrict__ x_out, float* __restrict__ segmax,
                       float* __restrict__ segsum, int nNodes) {
  int i = blockIdx.x * blockDim.x + threadIdx.x;
  int total = nNodes * ND;
  if (i < total) x_out[i] = 0.0f;
  if (i < nNodes) { segmax[i] = -INFINITY; segsum[i] = 0.0f; }
}

// att[e] = feat(e) . aw + ab   (fp32, full precision)
__global__ void k_att(const float* __restrict__ na, const float* __restrict__ ea,
                      const int* __restrict__ src, const int* __restrict__ dst,
                      const float* __restrict__ aw, const float* __restrict__ ab,
                      float* __restrict__ att, int nE) {
  int e = blockIdx.x * blockDim.x + threadIdx.x;
  if (e >= nE) return;
  float s = ab[0];
  const float4* a0 = (const float4*)(na + (size_t)src[e] * ND);
  const float4* a1 = (const float4*)(na + (size_t)dst[e] * ND);
  const float4* a2 = (const float4*)(ea + (size_t)e * ED);
  const float4* w0 = (const float4*)(aw);
  const float4* w1 = (const float4*)(aw + ND);
  const float4* w2 = (const float4*)(aw + 2 * ND);
#pragma unroll 4
  for (int k = 0; k < ND / 4; ++k) {
    float4 x = a0[k], w = w0[k];
    s += x.x * w.x + x.y * w.y + x.z * w.z + x.w * w.w;
  }
#pragma unroll 4
  for (int k = 0; k < ND / 4; ++k) {
    float4 x = a1[k], w = w1[k];
    s += x.x * w.x + x.y * w.y + x.z * w.z + x.w * w.w;
  }
#pragma unroll 4
  for (int k = 0; k < ED / 4; ++k) {
    float4 x = a2[k], w = w2[k];
    s += x.x * w.x + x.y * w.y + x.z * w.z + x.w * w.w;
  }
  att[e] = s;
}

__device__ __forceinline__ void atomicMaxF(float* addr, float v) {
  if (v >= 0.0f) atomicMax((int*)addr, __float_as_int(v));
  else           atomicMin((unsigned int*)addr, __float_as_uint(v));
}

__global__ void k_segmax(const int* __restrict__ src, const float* __restrict__ att,
                         float* __restrict__ segmax, int nE) {
  int e = blockIdx.x * blockDim.x + threadIdx.x;
  if (e < nE) atomicMaxF(&segmax[src[e]], att[e]);
}

__global__ void k_fixmax(float* __restrict__ segmax, int nNodes) {
  int i = blockIdx.x * blockDim.x + threadIdx.x;
  if (i < nNodes) { float s = segmax[i]; if (s < -3.0e38f) segmax[i] = 0.0f; }
}

__global__ void k_expsum(const int* __restrict__ src, const float* __restrict__ att,
                         const float* __restrict__ segmax, float* __restrict__ segsum,
                         int nE) {
  int e = blockIdx.x * blockDim.x + threadIdx.x;
  if (e < nE) {
    int s = src[e];
    atomicAdd(&segsum[s], __expf(att[e] - segmax[s]));
  }
}

// ---------------------------------------------------------------------------
// Fused main kernel: 32 edges per 256-thread block, both MLPs out of LDS
// ---------------------------------------------------------------------------
__global__ __launch_bounds__(256) void mpnn_main(
    const float* __restrict__ na, const float* __restrict__ ea,
    const int* __restrict__ e_src, const int* __restrict__ e_dst,
    const _Float16* __restrict__ wTm0, const float* __restrict__ mb0,
    const _Float16* __restrict__ wTm1, const float* __restrict__ mb1,
    const _Float16* __restrict__ wTm2, const float* __restrict__ mb2,
    const _Float16* __restrict__ wTe0, const float* __restrict__ eb0,
    const _Float16* __restrict__ wTe1, const float* __restrict__ eb1,
    const _Float16* __restrict__ wTe2, const float* __restrict__ eb2,
    const float* __restrict__ att, const float* __restrict__ segmax,
    const float* __restrict__ segsum,
    float* __restrict__ x_out, float* __restrict__ e_out, int nE_total)
{
  extern __shared__ char smem[];
  _Float16* feat = (_Float16*)smem;                 // 32 x 640 f16
  _Float16* hA   = feat + TILE_E * CATD;            // 32 x 256 f16
  _Float16* hB   = hA + TILE_E * ND;                // 32 x 256 f16
  int*   src_l   = (int*)(hB + TILE_E * ND);        // 32
  int*   dst_l   = src_l + TILE_E;                  // 32
  float* alpha_l = (float*)(dst_l + TILE_E);        // 32

  const int e0  = blockIdx.x * TILE_E;
  const int tid = threadIdx.x;
  const int nE  = min(TILE_E, nE_total - e0);

  if (tid < TILE_E) {
    if (tid < nE) {
      const int e = e0 + tid;
      const int s = e_src[e];
      src_l[tid] = s;
      dst_l[tid] = e_dst[e];
      const float den = segsum[s];
      alpha_l[tid] = (den > 0.0f) ? __expf(att[e] - segmax[s]) / den : 0.0f;
    } else {
      src_l[tid] = 0; dst_l[tid] = 0; alpha_l[tid] = 0.0f;
    }
  }
  __syncthreads();

  // Gather feat tile -> LDS f16 (zero-fill tail rows)
  for (int m = 0; m < TILE_E; ++m) {
    const bool ok = (m < nE);
    const int s = src_l[m], d = dst_l[m];
    for (int k = tid; k < CATD; k += 256) {
      float v = 0.0f;
      if (ok) {
        if (k < ND)            v = na[(size_t)s * ND + k];
        else if (k < 2 * ND)   v = na[(size_t)d * ND + (k - ND)];
        else                   v = ea[(size_t)(e0 + m) * ED + (k - 2 * ND)];
      }
      feat[(size_t)m * CATD + k] = (_Float16)v;
    }
  }
  __syncthreads();

  // ---- message MLP: 640 -> 256 -> 256 -> 256, last layer scatters alpha*msg
  wave_mlp_layer<0>(feat, CATD, wTm0, mb0, ND, hA, nullptr, nullptr, nullptr, nullptr, nE);
  __syncthreads();
  wave_mlp_layer<0>(hA, ND, wTm1, mb1, ND, hB, nullptr, nullptr, nullptr, nullptr, nE);
  __syncthreads();
  wave_mlp_layer<1>(hB, ND, wTm2, mb2, ND, nullptr, x_out, src_l, alpha_l, nullptr, nE);
  __syncthreads();

  // ---- edge MLP: 640 -> 128 -> 128 -> 128, last layer stores e_out
  wave_mlp_layer<0>(feat, CATD, wTe0, eb0, ED, hA, nullptr, nullptr, nullptr, nullptr, nE);
  __syncthreads();
  wave_mlp_layer<0>(hA, ED, wTe1, eb1, ED, hB, nullptr, nullptr, nullptr, nullptr, nE);
  __syncthreads();
  wave_mlp_layer<2>(hB, ED, wTe2, eb2, ED, nullptr, nullptr, nullptr, nullptr,
                    e_out + (size_t)e0 * ED, nE);
}

// ---------------------------------------------------------------------------
extern "C" void kernel_launch(void* const* d_in, const int* in_sizes, int n_in,
                              void* d_out, int out_size, void* d_ws, size_t ws_size,
                              hipStream_t stream) {
  const float* node_attr = (const float*)d_in[0];
  const float* edge_attr = (const float*)d_in[1];
  const int*   edge_idx  = (const int*)d_in[2];
  const float* mw0 = (const float*)d_in[3];  const float* mb0 = (const float*)d_in[4];
  const float* mw1 = (const float*)d_in[5];  const float* mb1 = (const float*)d_in[6];
  const float* mw2 = (const float*)d_in[7];  const float* mb2 = (const float*)d_in[8];
  const float* ew0 = (const float*)d_in[9];  const float* eb0 = (const float*)d_in[10];
  const float* ew1 = (const float*)d_in[11]; const float* eb1 = (const float*)d_in[12];
  const float* ew2 = (const float*)d_in[13]; const float* eb2 = (const float*)d_in[14];
  const float* aw  = (const float*)d_in[15]; const float* ab  = (const float*)d_in[16];

  const int nN = in_sizes[0] / ND;       // 10000
  const int nE = in_sizes[1] / ED;       // 320000
  const int* e_src = edge_idx;
  const int* e_dst = edge_idx + nE;

  // workspace partition
  char* ws = (char*)d_ws;
  _Float16* wTm0 = (_Float16*)ws; ws += (size_t)ND * CATD * 2;
  _Float16* wTm1 = (_Float16*)ws; ws += (size_t)ND * ND * 2;
  _Float16* wTm2 = (_Float16*)ws; ws += (size_t)ND * ND * 2;
  _Float16* wTe0 = (_Float16*)ws; ws += (size_t)ED * CATD * 2;
  _Float16* wTe1 = (_Float16*)ws; ws += (size_t)ED * ED * 2;
  _Float16* wTe2 = (_Float16*)ws; ws += (size_t)ED * ED * 2;
  float* att    = (float*)ws; ws += (size_t)nE * 4;
  float* segmax = (float*)ws; ws += (size_t)nN * 4;
  float* segsum = (float*)ws;

  float* x_out = (float*)d_out;
  float* e_out = x_out + (size_t)nN * ND;

  // 1) convert + transpose weights to f16 [N][K]
  auto cvt = [&](const float* w, _Float16* wT, int K, int N) {
    int total = K * N;
    k_wt<<<(total + 255) / 256, 256, 0, stream>>>(w, wT, K, N);
  };
  cvt(mw0, wTm0, CATD, ND); cvt(mw1, wTm1, ND, ND); cvt(mw2, wTm2, ND, ND);
  cvt(ew0, wTe0, CATD, ED); cvt(ew1, wTe1, ED, ED); cvt(ew2, wTe2, ED, ED);

  // 2) init x_out / segmax / segsum
  {
    int total = nN * ND;
    k_init<<<(total + 255) / 256, 256, 0, stream>>>(x_out, segmax, segsum, nN);
  }

  // 3) attention scores (fp32)
  k_att<<<(nE + 255) / 256, 256, 0, stream>>>(node_attr, edge_attr, e_src, e_dst,
                                              aw, ab, att, nE);
  // 4) scatter softmax stats
  k_segmax<<<(nE + 255) / 256, 256, 0, stream>>>(e_src, att, segmax, nE);
  k_fixmax<<<(nN + 255) / 256, 256, 0, stream>>>(segmax, nN);
  k_expsum<<<(nE + 255) / 256, 256, 0, stream>>>(e_src, att, segmax, segsum, nE);

  // 5) fused WMMA MLPs + scatter
  const size_t smem_bytes = (size_t)(TILE_E * CATD + 2 * TILE_E * ND) * 2
                          + (size_t)TILE_E * 4 * 3;  // 74112 B
  mpnn_main<<<(nE + TILE_E - 1) / TILE_E, 256, smem_bytes, stream>>>(
      node_attr, edge_attr, e_src, e_dst,
      wTm0, mb0, wTm1, mb1, wTm2, mb2,
      wTe0, eb0, wTe1, eb1, wTe2, eb2,
      att, segmax, segsum, x_out, e_out, nE);
}
